// LSTM_77756087927459
// MI455X (gfx1250) — matrix operand order repaired and checked by reading, not compiled, not run on hardware
//
#include <hip/hip_runtime.h>

// Problem constants (match reference)
#define Hd   1024
#define Bb   16
#define Tt   256
#define Vv   32000
#define MROWS (Bb * Tt)   // 4096

typedef __attribute__((ext_vector_type(16))) __bf16 v16bf;
typedef __attribute__((ext_vector_type(8)))  float  v8f;

union FragBF { v16bf v; float4 f4[2]; };

__device__ __forceinline__ __bf16 f2bf(float f) {
    unsigned u = __builtin_bit_cast(unsigned, f);
    unsigned r = (u + 0x7fffu + ((u >> 16) & 1u)) >> 16;   // round-to-nearest-even
    unsigned short s = (unsigned short)r;
    return __builtin_bit_cast(__bf16, s);
}

// CDNA5 async memory->LDS copy (ASYNCcnt-tracked, no VGPR data staging).
// the vdst VGPR gives the LDS byte address; low 32 bits of a flat shared pointer are the
// group-segment offset, which is exactly what VDST wants.
__device__ __forceinline__ void async_b128(unsigned lds_off, const void* gaddr) {
    asm volatile("global_load_async_to_lds_b128 %0, %1, off"
                 :: "v"(lds_off), "v"((unsigned long long)(size_t)gaddr)
                 : "memory");
}

// ---------------- prep kernels ----------------

// src is [K][N] row-major fp32 ; dst is [N][K] row-major bf16 (transposed)
__global__ void k_transpose_bf16(const float* __restrict__ src,
                                 __bf16* __restrict__ dst, int K, int N) {
    long long idx = (long long)blockIdx.x * blockDim.x + threadIdx.x;
    if (idx >= (long long)K * N) return;
    int k = (int)(idx / N);
    int n = (int)(idx % N);
    dst[(long long)n * K + k] = f2bf(src[idx]);
}

// x_bf16[m][h] = bf16(emb[tokens[m]][h]), m = b*T + t
__global__ void k_embed(const int* __restrict__ tokens,
                        const float* __restrict__ emb,
                        __bf16* __restrict__ x) {
    int idx = blockIdx.x * blockDim.x + threadIdx.x;   // 4096*1024 threads
    int m = idx >> 10;
    int h = idx & (Hd - 1);
    int tok = tokens[m];
    x[idx] = f2bf(emb[(long long)tok * Hd + h]);
}

__global__ void k_init_state(__bf16* __restrict__ hcur, float* __restrict__ c) {
    int idx = blockIdx.x * blockDim.x + threadIdx.x;   // 16*1024 threads
    hcur[idx] = f2bf(0.f);
    c[idx] = 0.f;
}

// ---------------- generic WMMA GEMM ----------------
// C[M][N] = A[M][K] (bf16, row-major) * BT[N][K]^T (bf16, pre-transposed) + bias[N]
// 256 threads = 8 waves; 128x128 tile; wave grid 2(M) x 4(N); 4x2 WMMA tiles/wave.
// Double-buffered LDS staged with global_load_async_to_lds_b128; epilogue-peeled
// steady state (prefetch s+1, s_wait_asynccnt 4, compute s).
// LDS row stride 40 halves (80 B): 16B aligned for b128, conflict-free frag reads.
// NT=true: non-temporal result stores (stream 524 MB of logits past L2 so the
// L2-resident bf16 w_out stays hot).
template <bool NT>
__global__ __launch_bounds__(256) void k_gemm(const __bf16* __restrict__ A,
                                              const __bf16* __restrict__ BT,
                                              const float* __restrict__ bias,
                                              float* __restrict__ C,
                                              int M, int N, int K) {
    __shared__ __bf16 lds[2 * 2 * 128 * 40];   // [buf][A|B][128][40] = 40 KB

    const int tid  = threadIdx.x;
    const int lane = tid & 31;
    const int w    = tid >> 5;
    const int waveM = w >> 2;        // 0..1
    const int waveN = w & 3;         // 0..3
    const int m0 = blockIdx.y * 128;
    const int n0 = blockIdx.x * 128;
    const int hi = lane >> 4;
    const int lm = lane & 15;

    // staging assignment: thread -> chunks (row, j*8) and (row+64, j*8), 16 B each
    const int row = tid >> 2;        // 0..63
    const int j   = tid & 3;         // 0..3

    const char* gA0 = (const char*)(A  + (long long)(m0 + row)      * K + j * 8);
    const char* gA1 = (const char*)(A  + (long long)(m0 + row + 64) * K + j * 8);
    const char* gB0 = (const char*)(BT + (long long)(n0 + row)      * K + j * 8);
    const char* gB1 = (const char*)(BT + (long long)(n0 + row + 64) * K + j * 8);

    const unsigned TILE = 128u * 40u * 2u;                 // bytes per tile
    const unsigned ldsBase = (unsigned)(size_t)&lds[0];    // group-segment offset
    const unsigned dA0 = ldsBase + (unsigned)(row * 40 + j * 8) * 2u;
    const unsigned dA1 = dA0 + 64u * 40u * 2u;
    const unsigned dB0 = dA0 + TILE;
    const unsigned dB1 = dA1 + TILE;

    const v8f vzero = {};
    v8f acc[4][2];
    for (int i = 0; i < 4; i++)
        for (int jj = 0; jj < 2; jj++) acc[i][jj] = vzero;

    // compute one 32-deep K stage from LDS buffer `buf`:
    // all 12 fragment ds_load_b128 first, then an uninterrupted 8-WMMA chain.
    auto compute = [&](int buf) {
        const __bf16* tA = &lds[(size_t)buf * 2u * 128u * 40u];
        const __bf16* tB = tA + 128 * 40;
        FragBF bfr[2], afr[4];
        for (int nt = 0; nt < 2; nt++) {
            // B fragment: lane holds column n, contiguous K = hi*16 .. hi*16+15
            const __bf16* p = &tB[(waveN * 32 + nt * 16 + lm) * 40 + hi * 16];
            bfr[nt].f4[0] = *(const float4*)(p);
            bfr[nt].f4[1] = *(const float4*)(p + 8);
        }
        for (int mt = 0; mt < 4; mt++) {
            // A fragment: lane holds row m, contiguous K = hi*16 .. hi*16+15
            const __bf16* p = &tA[(waveM * 64 + mt * 16 + lm) * 40 + hi * 16];
            afr[mt].f4[0] = *(const float4*)(p);
            afr[mt].f4[1] = *(const float4*)(p + 8);
        }
        for (int mt = 0; mt < 4; mt++)
            for (int nt = 0; nt < 2; nt++)
                acc[mt][nt] = __builtin_amdgcn_wmma_f32_16x16x32_bf16(
                    false, afr[mt].v, false, bfr[nt].v, (short)0, acc[mt][nt],
                    false, false);
    };

    const int S = K >> 5;            // K stages of 32

    // prologue: stage 0 -> buffer 0
    async_b128(dA0, gA0);
    async_b128(dA1, gA1);
    async_b128(dB0, gB0);
    async_b128(dB1, gB1);

    // steady state (branch-free body): prefetch s+1, wait for s, compute s
    for (int s = 0; s < S - 1; s++) {
        const unsigned boff = (unsigned)((s + 1) & 1) * (2u * TILE);
        const size_t go = (size_t)(s + 1) * 64;            // 32 halves per stage
        async_b128(dA0 + boff, gA0 + go);
        async_b128(dA1 + boff, gA1 + go);
        async_b128(dB0 + boff, gB0 + go);
        async_b128(dB1 + boff, gB1 + go);
        asm volatile("s_wait_asynccnt 4" ::: "memory");    // stage s landed
        __syncthreads();
        compute(s & 1);
        __syncthreads();   // all reads of buf done before stage s+2 overwrites it
    }
    // epilogue: last stage
    asm volatile("s_wait_asynccnt 0" ::: "memory");
    __syncthreads();
    compute((S - 1) & 1);

    // store: C/D layout -> lane half selects M offset (hi*8 + r), N = lane&15
    for (int mt = 0; mt < 4; mt++) {
        int rowBase = m0 + waveM * 64 + mt * 16 + hi * 8;
        for (int nt = 0; nt < 2; nt++) {
            int col = n0 + waveN * 32 + nt * 16 + lm;
            float bv = bias ? bias[col] : 0.f;
            for (int r = 0; r < 8; r++) {
                float val = acc[mt][nt][r] + bv;
                float* dst = &C[(long long)(rowBase + r) * N + col];
                if (NT) __builtin_nontemporal_store(val, dst);
                else    *dst = val;
            }
        }
    }
}

// ---------------- recurrence ----------------
// One wave per (gate, n-tile): M=16 (batch) fits exactly one WMMA C tile.
// gates[g][b][n] = xg[g][b*T+t][n] + sum_k h[b][k] * u_g[k][n]
__global__ __launch_bounds__(256) void k_rec_gemm(const __bf16* __restrict__ hcur,
                                                  const __bf16* __restrict__ uT,
                                                  const float* __restrict__ xg,
                                                  float* __restrict__ gates, int t) {
    const int tid  = threadIdx.x;
    const int lane = tid & 31;
    const int w    = tid >> 5;
    const int idx  = blockIdx.x * 8 + w;     // 0..255
    const int gate  = idx >> 6;
    const int ntile = idx & 63;
    const int hi = lane >> 4;
    const int lm = lane & 15;
    const __bf16* uTg = uT + (long long)gate * Hd * Hd;

    const __bf16* pa0 = hcur + lm * Hd + hi * 16;
    const __bf16* pb0 = uTg + (long long)(ntile * 16 + lm) * Hd + hi * 16;

    v8f acc = {};
    for (int kk = 0; kk < Hd; kk += 32) {
        FragBF a, b;
        a.f4[0] = *(const float4*)(pa0 + kk);
        a.f4[1] = *(const float4*)(pa0 + kk + 8);
        b.f4[0] = *(const float4*)(pb0 + kk);
        b.f4[1] = *(const float4*)(pb0 + kk + 8);
        acc = __builtin_amdgcn_wmma_f32_16x16x32_bf16(
            false, a.v, false, b.v, (short)0, acc, false, false);
    }
    int col = ntile * 16 + lm;
    for (int r = 0; r < 8; r++) {
        int bRow = hi * 8 + r;               // batch index
        float pre = xg[((long long)gate * MROWS + (bRow * Tt + t)) * Hd + col];
        gates[(gate * Bb + bRow) * Hd + col] = acc[r] + pre;
    }
}

__global__ void k_rec_update(const float* __restrict__ gates,
                             float* __restrict__ c,
                             __bf16* __restrict__ hcur,
                             __bf16* __restrict__ hs, int t) {
    int idx = blockIdx.x * blockDim.x + threadIdx.x;   // 16*1024 threads
    int b = idx >> 10;
    int n = idx & (Hd - 1);
    float pf = gates[(0 * Bb + b) * Hd + n];
    float pi = gates[(1 * Bb + b) * Hd + n];
    float pc = gates[(2 * Bb + b) * Hd + n];
    float po = gates[(3 * Bb + b) * Hd + n];
    float f = 1.f / (1.f + __expf(-pf));
    float i = 1.f / (1.f + __expf(-pi));
    float g = tanhf(pc);
    float o = 1.f / (1.f + __expf(-po));
    float cn = f * c[idx] + i * g;
    c[idx] = cn;
    float h = o * tanhf(cn);
    hcur[idx] = f2bf(h);
    hs[(long long)(b * Tt + t) * Hd + n] = f2bf(h);
}

// ---------------- host ----------------

extern "C" void kernel_launch(void* const* d_in, const int* in_sizes, int n_in,
                              void* d_out, int out_size, void* d_ws, size_t ws_size,
                              hipStream_t stream) {
    const int*   tokens = (const int*)d_in[0];
    const float* emb    = (const float*)d_in[1];
    const float* w[4]   = {(const float*)d_in[2],  (const float*)d_in[5],
                           (const float*)d_in[8],  (const float*)d_in[11]};
    const float* u[4]   = {(const float*)d_in[3],  (const float*)d_in[6],
                           (const float*)d_in[9],  (const float*)d_in[12]};
    const float* bg[4]  = {(const float*)d_in[4],  (const float*)d_in[7],
                           (const float*)d_in[10], (const float*)d_in[13]};
    const float* w_out  = (const float*)d_in[14];
    const float* b_out  = (const float*)d_in[15];
    float* out = (float*)d_out;

    // workspace carve-up (256B aligned; ~167 MB total)
    char* ws = (char*)d_ws;
    size_t off = 0;
    auto carve = [&](size_t bytes) {
        void* p = ws + off;
        off = (off + bytes + 255) & ~(size_t)255;
        return p;
    };
    __bf16* xbf   = (__bf16*)carve((size_t)MROWS * Hd * 2);      //  8 MB
    __bf16* wT    = (__bf16*)carve((size_t)4 * Hd * Hd * 2);     //  8 MB
    __bf16* uT    = (__bf16*)carve((size_t)4 * Hd * Hd * 2);     //  8 MB
    __bf16* woutT = (__bf16*)carve((size_t)Vv * Hd * 2);         // 64 MB (L2-resident)
    float*  xg    = (float*)carve((size_t)4 * MROWS * Hd * 4);   // 64 MB
    __bf16* hsbuf = (__bf16*)carve((size_t)MROWS * Hd * 2);      //  8 MB
    __bf16* hcur  = (__bf16*)carve((size_t)Bb * Hd * 2);
    float*  cst   = (float*)carve((size_t)Bb * Hd * 4);
    float*  gates = (float*)carve((size_t)4 * Bb * Hd * 4);
    (void)ws_size; (void)in_sizes; (void)n_in; (void)out_size;

    // 1) weight transpose + bf16 convert
    for (int g = 0; g < 4; g++) {
        k_transpose_bf16<<<(Hd * Hd + 255) / 256, 256, 0, stream>>>(
            w[g], wT + (long long)g * Hd * Hd, Hd, Hd);
        k_transpose_bf16<<<(Hd * Hd + 255) / 256, 256, 0, stream>>>(
            u[g], uT + (long long)g * Hd * Hd, Hd, Hd);
    }
    k_transpose_bf16<<<(int)(((long long)Hd * Vv + 255) / 256), 256, 0, stream>>>(
        w_out, woutT, Hd, Vv);

    // 2) embedding gather (fp32 -> bf16)
    k_embed<<<MROWS * Hd / 256, 256, 0, stream>>>(tokens, emb, xbf);

    // 3) input-gate GEMMs: xg[g] = x @ w_g + b_g   (4096x1024x1024, WMMA)
    for (int g = 0; g < 4; g++) {
        k_gemm<false><<<dim3(Hd / 128, MROWS / 128), 256, 0, stream>>>(
            xbf, wT + (long long)g * Hd * Hd, bg[g],
            xg + (long long)g * MROWS * Hd, MROWS, Hd, Hd);
    }

    // 4) recurrence over T
    k_init_state<<<Bb * Hd / 256, 256, 0, stream>>>(hcur, cst);
    for (int t = 0; t < Tt; t++) {
        k_rec_gemm<<<32, 256, 0, stream>>>(hcur, uT, xg, gates, t);
        k_rec_update<<<Bb * Hd / 256, 256, 0, stream>>>(gates, cst, hcur, hsbuf, t);
    }

    // 5) output projection: logits = hs @ w_out + b_out (4096x32000x1024, WMMA)
    //    NT stores: stream logits past L2, keep w_outT resident.
    k_gemm<true><<<dim3(Vv / 128, MROWS / 128), 256, 0, stream>>>(
        hsbuf, woutT, b_out, out, MROWS, Vv, Hd);
}
